// FARGANSub_46806553592417
// MI455X (gfx1250) — matrix operand order repaired and checked by reading, not compiled, not run on hardware
//
#include <hip/hip_runtime.h>
#include <hip/hip_bf16.h>

typedef __attribute__((ext_vector_type(16))) _Float16 v16h;
typedef __attribute__((ext_vector_type(8)))  float    v8f;

// ---------------- converted-weight layout in d_ws (f16 elements) ----------------
// all stored row-major [out][Kpad]
#define SZ_FWC0    (256*736)
#define SZ_256     (256*256)
#define SZ_GRUW    (768*256)
#define SZ_SIG     (48*1024)
#define OFF_FWC0    0
#define OFF_FWC0GLU (OFF_FWC0    + SZ_FWC0)
#define OFF_D2      (OFF_FWC0GLU + SZ_256)
#define OFF_D2GLU   (OFF_D2      + SZ_256)
#define OFF_G1WIH   (OFF_D2GLU   + SZ_256)
#define OFF_G1WHH   (OFF_G1WIH   + SZ_GRUW)
#define OFF_G1GLU   (OFF_G1WHH   + SZ_GRUW)
#define OFF_G2WIH   (OFF_G1GLU   + SZ_256)
#define OFF_G2WHH   (OFF_G2WIH   + SZ_GRUW)
#define OFF_G2GLU   (OFF_G2WHH   + SZ_GRUW)
#define OFF_G3WIH   (OFF_G2GLU   + SZ_256)
#define OFF_G3WHH   (OFF_G3WIH   + SZ_GRUW)
#define OFF_G3GLU   (OFF_G3WHH   + SZ_GRUW)
#define OFF_SIG     (OFF_G3GLU   + SZ_256)
#define WS_HALFS    (OFF_SIG     + SZ_SIG)

// ---------------- fp32 -> f16 weight conversion (zero-padded) ----------------
__global__ void cvt_w_kernel(const float* __restrict__ src, _Float16* __restrict__ dst,
                             int outf, int inf, int outp, int inp) {
  int i = blockIdx.x * 256 + threadIdx.x;
  int tot = outp * inp;
  if (i >= tot) return;
  int r = i / inp, c = i - r * inp;
  float v = (r < outf && c < inf) ? src[r * inf + c] : 0.f;
  dst[i] = (_Float16)v;
}

// ---------------- WMMA fragment helpers (layouts per ISA 7.12.2) ----------------
// A: 16x32 f16. lanes 0-15: M=lane, K = {0..7,16..23}; lanes 16-31: M=lane-16, K={8..15,24..31}
__device__ __forceinline__ v16h load_a(const _Float16* __restrict__ A, int lda, int kbase) {
  int lane = threadIdx.x & 31;
  int m = lane & 15;
  int hh = lane >> 4;
  union { v16h v; _Float16 h[16]; } u;
#pragma unroll
  for (int r = 0; r < 8; ++r) {
    int k = kbase + ((r >> 2) << 4) + hh * 8 + (r & 3) * 2;
    u.h[2 * r]     = A[m * lda + k];
    u.h[2 * r + 1] = A[m * lda + k + 1];
  }
  return u.v;
}

// B: 32x16 f16 (weight row n supplies column n). lane&15 = N; lanes 0-15: K={0..7,16..23}; lanes 16-31: K={8..15,24..31}
__device__ __forceinline__ v16h load_b(const _Float16* __restrict__ W, int ldw, int n0, int kbase) {
  int lane = threadIdx.x & 31;
  int n = lane & 15;
  int hh = lane >> 4;
  union { v16h v; _Float16 h[16]; } u;
#pragma unroll
  for (int r = 0; r < 8; ++r) {
    int k = kbase + ((r >> 2) << 4) + hh * 8 + (r & 3) * 2;
    u.h[2 * r]     = W[(n0 + n) * ldw + k];
    u.h[2 * r + 1] = W[(n0 + n) * ldw + k + 1];
  }
  return u.v;
}

__device__ __forceinline__ v8f wmma_acc(v16h a, v16h b, v8f c) {
  return __builtin_amdgcn_wmma_f32_16x16x32_f16(false, a, false, b, (short)0, c, false, false);
}

#define V8F_ZERO {0.f, 0.f, 0.f, 0.f, 0.f, 0.f, 0.f, 0.f}

// single-tile GEMM (sig projection)
__device__ __forceinline__ v8f wmma_gemm(const _Float16* __restrict__ A, int lda, int aoff,
                                         const _Float16* __restrict__ W, int ldw, int n0,
                                         int ktiles) {
  v8f acc = V8F_ZERO;
  for (int kt = 0; kt < ktiles; ++kt) {
    v16h a = load_a(A + aoff, lda, kt * 32);
    acc = wmma_acc(a, load_b(W, ldw, n0, kt * 32), acc);
  }
  return acc;
}

// branch-free transcendentals: v_exp_f32 + v_rcp_f32, no libm control flow
__device__ __forceinline__ float sigm(float x) {
  return __builtin_amdgcn_rcpf(1.f + __expf(-x));
}
__device__ __forceinline__ float fast_tanh(float x) {
  // tanh(x) = 1 - 2/(exp(2x)+1); saturates correctly at +-1 (exp->inf => rcp->0)
  return 1.f - 2.f * __builtin_amdgcn_rcpf(1.f + __expf(2.f * x));
}

struct KParams {
  const float* cond; const float* prev; const float* exc_mem; const float* phase;
  const int* period;
  const float* state0; const float* state1; const float* state2; const float* state3;
  const float* gain; const float* pgain_w; const float* pgain_b;
  const _Float16* w;   // converted weights base
  float* o_sig; float* o_exc; float* o_g1; float* o_g2; float* o_g3; float* o_fws;
};

// dynamic LDS carve (bytes)
#define L_P0    0          // f32 16x256
#define L_P1    16384      // f32 16x256
#define L_H0    32768      // f16 16x256
#define L_H1    40960      // f16 16x256
#define L_FEAT  49152      // f16 16x1024
#define L_XCAT  81920      // f16 16x736
#define L_FP    105472     // f32 16x40
#define L_GN    108032     // f32 16
#define L_PG    108096     // f32 16
#define L_SB    108160     // f32 16x48
#define LDS_BYTES 111232

__device__ __forceinline__ float xcat_val(const KParams& p, int row, int c, float inv, int idx0) {
  if (c < 480) return p.state3[(size_t)row * 480 + c];
  int tc = c - 480;
  if (tc < 80)  return p.cond[row * 80 + tc];
  if (tc < 120) return p.exc_mem[row * 256 + idx0 + 2 + (tc - 80)] * inv;
  if (tc < 160) return p.prev[row * 40 + (tc - 120)] * inv;
  return p.phase[row * 80 + (tc - 160)];
}

__device__ __forceinline__ void load_state_lds(const float* st, int rowBase, float* Pf, _Float16* Hf) {
  for (int e = threadIdx.x; e < 4096; e += 256) {
    int r = e >> 8, c = e & 255;
    float v = st[(size_t)(rowBase + r) * 256 + c];
    Pf[e] = v;
    Hf[e] = (_Float16)v;
  }
}

// tanh(A @ W.T), 256-wide output; each wave owns N-tiles {wv*16, wv*16+128},
// sharing one A-fragment load per k-step across both tiles.
__device__ __forceinline__ void tanh_stage(const _Float16* A, int lda, int aoff,
                                           const _Float16* W, int ldw, int ktiles,
                                           float* Pdst, _Float16* Hdst) {
  int wv = threadIdx.x >> 5, lane = threadIdx.x & 31;
  int n = lane & 15, hh = lane >> 4;
  int n0 = wv * 16, n1 = n0 + 128;
  v8f a0 = V8F_ZERO, a1 = V8F_ZERO;
  for (int kt = 0; kt < ktiles; ++kt) {
    v16h a = load_a(A + aoff, lda, kt * 32);
    a0 = wmma_acc(a, load_b(W, ldw, n0, kt * 32), a0);
    a1 = wmma_acc(a, load_b(W, ldw, n1, kt * 32), a1);
  }
#pragma unroll
  for (int r = 0; r < 8; ++r) {
    int m = r + hh * 8;
    float v0 = fast_tanh(a0[r]);
    float v1 = fast_tanh(a1[r]);
    Pdst[m * 256 + n0 + n] = v0;  Hdst[m * 256 + n0 + n] = (_Float16)v0;
    Pdst[m * 256 + n1 + n] = v1;  Hdst[m * 256 + n1 + n] = (_Float16)v1;
  }
}

// x * sigmoid(x @ W.T): x f32 in Px / f16 in X16 -> f16 dst (ld/off for feat slices)
__device__ __forceinline__ void glu_stage(const float* Px, const _Float16* X16,
                                          const _Float16* Wglu,
                                          _Float16* dst, int ldd, int doff) {
  int wv = threadIdx.x >> 5, lane = threadIdx.x & 31;
  int n = lane & 15, hh = lane >> 4;
  int n0 = wv * 16, n1 = n0 + 128;
  v8f a0 = V8F_ZERO, a1 = V8F_ZERO;
  for (int kt = 0; kt < 8; ++kt) {
    v16h a = load_a(X16, 256, kt * 32);
    a0 = wmma_acc(a, load_b(Wglu, 256, n0, kt * 32), a0);
    a1 = wmma_acc(a, load_b(Wglu, 256, n1, kt * 32), a1);
  }
#pragma unroll
  for (int r = 0; r < 8; ++r) {
    int m = r + hh * 8;
    dst[m * ldd + doff + n0 + n] = (_Float16)(Px[m * 256 + n0 + n] * sigm(a0[r]));
    dst[m * ldd + doff + n1 + n] = (_Float16)(Px[m * 256 + n1 + n] * sigm(a1[r]));
  }
}

// full GRU cell: per 16x16 output tile, one k-loop feeds all 6 gate accumulators
// from shared a_x / a_h fragment loads.
__device__ __forceinline__ void gru_stage(const _Float16* Ax, int lda, int aoff,
                                          const _Float16* Hh, const float* Ph,
                                          const _Float16* Wih, const _Float16* Whh,
                                          float* outState, int rowBase,
                                          float* Pdst, _Float16* Hdst) {
  int wv = threadIdx.x >> 5, lane = threadIdx.x & 31;
  int n = lane & 15, hh = lane >> 4;
  for (int t = wv; t < 16; t += 8) {
    int n0 = t * 16;
    v8f ir = V8F_ZERO, iz = V8F_ZERO, in = V8F_ZERO;
    v8f hr = V8F_ZERO, hz = V8F_ZERO, hn = V8F_ZERO;
    for (int kt = 0; kt < 8; ++kt) {
      int kb = kt * 32;
      v16h ax = load_a(Ax + aoff, lda, kb);
      v16h ah = load_a(Hh, 256, kb);
      ir = wmma_acc(ax, load_b(Wih, 256, n0,       kb), ir);
      iz = wmma_acc(ax, load_b(Wih, 256, n0 + 256, kb), iz);
      in = wmma_acc(ax, load_b(Wih, 256, n0 + 512, kb), in);
      hr = wmma_acc(ah, load_b(Whh, 256, n0,       kb), hr);
      hz = wmma_acc(ah, load_b(Whh, 256, n0 + 256, kb), hz);
      hn = wmma_acc(ah, load_b(Whh, 256, n0 + 512, kb), hn);
    }
#pragma unroll
    for (int r = 0; r < 8; ++r) {
      int m = r + hh * 8;
      float rg = sigm(ir[r] + hr[r]);
      float zg = sigm(iz[r] + hz[r]);
      float ng = fast_tanh(in[r] + rg * hn[r]);
      float ho = Ph[m * 256 + n0 + n];
      float hv = (1.f - zg) * ng + zg * ho;
      outState[(size_t)(rowBase + m) * 256 + n0 + n] = hv;
      Pdst[m * 256 + n0 + n] = hv;
      Hdst[m * 256 + n0 + n] = (_Float16)hv;
    }
  }
}

__global__ __launch_bounds__(256) void fargan_fused_kernel(KParams p) {
  extern __shared__ char smem[];
  float*    P0   = (float*)(smem + L_P0);
  float*    P1   = (float*)(smem + L_P1);
  _Float16* H0   = (_Float16*)(smem + L_H0);
  _Float16* H1   = (_Float16*)(smem + L_H1);
  _Float16* feat = (_Float16*)(smem + L_FEAT);
  _Float16* xcat = (_Float16*)(smem + L_XCAT);
  float*    fpit = (float*)(smem + L_FP);
  float*    gnS  = (float*)(smem + L_GN);
  float*    pgS  = (float*)(smem + L_PG);
  float*    sgb  = (float*)(smem + L_SB);

  const int rowBase = blockIdx.x * 16;
  const int tid = threadIdx.x;
  const int wv = tid >> 5, lane = tid & 31;
  const int cn = lane & 15, hh = lane >> 4;

  // ---- S1: gather + build xcat (f16, K padded to 736), fwc0_state out, exc_mem copy
  {
    int rid = tid >> 4, sub = tid & 15;
    int row = rowBase + rid;
    float g = p.gain[row];
    float inv = __builtin_amdgcn_rcpf(1e-5f + g);
    int per = p.period[row];
    int pc = per < 42 ? 42 : (per > 254 ? 254 : per);
    int idx0 = 254 - pc;
    if (sub == 0) gnS[rid] = g;
    for (int c = sub; c < 736; c += 16)
      xcat[rid * 736 + c] = (_Float16)((c < 720) ? xcat_val(p, row, c, inv, idx0) : 0.f);
    for (int c = sub; c < 480; c += 16)
      p.o_fws[(size_t)row * 480 + c] = xcat_val(p, row, 240 + c, inv, idx0);
    for (int j = sub; j < 40; j += 16)
      fpit[rid * 40 + j] = p.exc_mem[row * 256 + idx0 + 2 + j] * inv;
    for (int c = sub; c < 216; c += 16)
      p.o_exc[(size_t)row * 256 + c] = p.exc_mem[row * 256 + 40 + c];
  }
  __syncthreads();

  // ---- S2: fwc0_pre = tanh(xcat @ fwc0_w.T)   K=736
  tanh_stage(xcat, 736, 0, p.w + OFF_FWC0, 736, 23, P0, H0);
  __syncthreads();
  // ---- S3: fwc0_out = glu(fwc0_pre)
  glu_stage(P0, H0, p.w + OFF_FWC0GLU, H1, 256, 0);
  __syncthreads();
  // ---- S4: t = tanh(fwc0_out @ d2_w.T)
  tanh_stage(H1, 256, 0, p.w + OFF_D2, 256, 8, P0, H0);
  __syncthreads();
  // ---- S5: dense2_out = glu(t) -> feat[:,768:1024]; preload state0
  glu_stage(P0, H0, p.w + OFF_D2GLU, feat, 1024, 768);
  load_state_lds(p.state0, rowBase, P1, H1);
  __syncthreads();
  // ---- S6: GRU1
  gru_stage(feat, 1024, 768, H1, P1, p.w + OFF_G1WIH, p.w + OFF_G1WHH, p.o_g1, rowBase, P0, H0);
  __syncthreads();
  // ---- S7: gru1_out -> feat[:,0:256]; preload state1
  glu_stage(P0, H0, p.w + OFF_G1GLU, feat, 1024, 0);
  load_state_lds(p.state1, rowBase, P1, H1);
  __syncthreads();
  // ---- S8: GRU2
  gru_stage(feat, 1024, 0, H1, P1, p.w + OFF_G2WIH, p.w + OFF_G2WHH, p.o_g2, rowBase, P0, H0);
  __syncthreads();
  // ---- S9: gru2_out -> feat[:,256:512]; preload state2
  glu_stage(P0, H0, p.w + OFF_G2GLU, feat, 1024, 256);
  load_state_lds(p.state2, rowBase, P1, H1);
  __syncthreads();
  // ---- S10: GRU3
  gru_stage(feat, 1024, 256, H1, P1, p.w + OFF_G3WIH, p.w + OFF_G3WHH, p.o_g3, rowBase, P0, H0);
  __syncthreads();
  // ---- S11: gru3_out -> feat[:,512:768]
  glu_stage(P0, H0, p.w + OFF_G3GLU, feat, 1024, 512);
  __syncthreads();
  // ---- S12: sig = tanh(feat @ sig_out_w.T) (N padded 40->48); pgain = exp(feat . pg_w + b)
  if (wv < 3) {
    int n0 = wv * 16;
    v8f acc = wmma_gemm(feat, 1024, 0, p.w + OFF_SIG, 1024, n0, 32);
#pragma unroll
    for (int r = 0; r < 8; ++r) {
      int m = r + hh * 8;
      sgb[m * 48 + n0 + cn] = fast_tanh(acc[r]);
    }
  } else if (wv == 3 && lane < 16) {
    int r = lane;
    float s = 0.f;
    for (int k = 0; k < 1024; ++k) s += (float)feat[r * 1024 + k] * p.pgain_w[k];
    pgS[r] = __expf(s + p.pgain_b[0]);
  }
  __syncthreads();
  // ---- S13: sig_out = (sig + pgain*fpitch)*gain; finish exc_mem_new tail
  for (int e = tid; e < 640; e += 256) {
    int r = e / 40, c = e - r * 40;
    int row = rowBase + r;
    float v = (sgb[r * 48 + c] + pgS[r] * fpit[r * 40 + c]) * gnS[r];
    p.o_sig[(size_t)row * 40 + c] = v;
    p.o_exc[(size_t)row * 256 + 216 + c] = v;
  }
}

extern "C" void kernel_launch(void* const* d_in, const int* in_sizes, int n_in,
                              void* d_out, int out_size, void* d_ws, size_t ws_size,
                              hipStream_t stream) {
  const int B = in_sizes[0] / 80;
  _Float16* w = (_Float16*)d_ws;

  struct CvtDesc { int src_idx; int outf, inf, outp, inp; long off; };
  const CvtDesc cv[] = {
    {10, 256, 720, 256, 736, OFF_FWC0},
    {11, 256, 256, 256, 256, OFF_FWC0GLU},
    {12, 256, 256, 256, 256, OFF_D2},
    {13, 256, 256, 256, 256, OFF_D2GLU},
    {14, 768, 256, 768, 256, OFF_G1WIH},
    {15, 768, 256, 768, 256, OFF_G1WHH},
    {16, 256, 256, 256, 256, OFF_G1GLU},
    {17, 768, 256, 768, 256, OFF_G2WIH},
    {18, 768, 256, 768, 256, OFF_G2WHH},
    {19, 256, 256, 256, 256, OFF_G2GLU},
    {20, 768, 256, 768, 256, OFF_G3WIH},
    {21, 768, 256, 768, 256, OFF_G3WHH},
    {22, 256, 256, 256, 256, OFF_G3GLU},
    {23,  40, 1024, 48, 1024, OFF_SIG},
  };
  for (const auto& c : cv) {
    int tot = c.outp * c.inp;
    cvt_w_kernel<<<(tot + 255) / 256, 256, 0, stream>>>(
        (const float*)d_in[c.src_idx], w + c.off, c.outf, c.inf, c.outp, c.inp);
  }

  float* out = (float*)d_out;
  KParams p;
  p.cond    = (const float*)d_in[0];
  p.prev    = (const float*)d_in[1];
  p.exc_mem = (const float*)d_in[2];
  p.phase   = (const float*)d_in[3];
  p.period  = (const int*)d_in[4];
  p.state0  = (const float*)d_in[5];
  p.state1  = (const float*)d_in[6];
  p.state2  = (const float*)d_in[7];
  p.state3  = (const float*)d_in[8];
  p.gain    = (const float*)d_in[9];
  p.pgain_w = (const float*)d_in[24];
  p.pgain_b = (const float*)d_in[25];
  p.w = w;
  size_t o = 0;
  p.o_sig = out;                 o += (size_t)B * 40;
  p.o_exc = out + o;             o += (size_t)B * 256;
  p.o_g1  = out + o;             o += (size_t)B * 256;
  p.o_g2  = out + o;             o += (size_t)B * 256;
  p.o_g3  = out + o;             o += (size_t)B * 256;
  p.o_fws = out + o;

  fargan_fused_kernel<<<B / 16, 256, LDS_BYTES, stream>>>(p);
}
